// ClassifierHead_44753559224525
// MI455X (gfx1250) — matrix operand order repaired and checked by reading, not compiled, not run on hardware
//
#include <hip/hip_runtime.h>
#include <hip/hip_bf16.h>
#include <cstdint>

#define NTAG 9
#define SEQ  2048
#define BATCH 16
#define HID  1024
#define NEGC (-10000.0f)

typedef __attribute__((ext_vector_type(16))) __bf16 v16bf;
typedef __attribute__((ext_vector_type(8)))  float  v8f;

union BFrag { v16bf v; uint4 q[2]; unsigned short us[16]; };

#if __has_builtin(__builtin_amdgcn_global_load_async_to_lds_b128)
#define USE_ASYNC 1
// Builtin parameter type per hipcc diagnostic: pointer to int __vector(4),
// in AS1 (global) / AS3 (LDS).
typedef int async_v4i __attribute__((vector_size(16)));
typedef __attribute__((address_space(1))) async_v4i* async_gptr;
typedef __attribute__((address_space(3))) async_v4i* async_lptr;
#else
#define USE_ASYNC 0
#endif

#if __has_builtin(__builtin_amdgcn_s_wait_asynccnt)
#define WAIT_ASYNC(n) do { __builtin_amdgcn_s_wait_asynccnt(n); \
                           asm volatile("" ::: "memory"); } while (0)
#else
#define WAIT_ASYNC(n) asm volatile("s_wait_asynccnt %0" :: "n"(n) : "memory")
#endif

__device__ __forceinline__ float lrelu(float x) {
    return fmaxf(x, 0.0f) + 0.01f * fminf(x, 0.0f);
}

__device__ __forceinline__ unsigned short f2bf(float f) {
    union { __bf16 b[2]; unsigned short u[2]; } cv;
    cv.b[0] = (__bf16)f;           // native f32->bf16 convert
    return cv.u[0];
}

// ---------------------------------------------------------------------------
// Kernel 1: repack W (1024x9 f32) into per-lane bf16 B-fragments.
// B packing chosen here (A uses the same k order): lane n=lane&15 is column; khalf=lane>>4;
// the 16 halves per lane are K = kbase + 16*khalf + (0..15), contiguous.
// ---------------------------------------------------------------------------
__global__ void pack_w_kernel(const float* __restrict__ W,
                              unsigned short* __restrict__ wpack) {
    int c = blockIdx.x;          // k-chunk 0..31
    int lane = threadIdx.x;      // 0..31
    int n = lane & 15;
    int khalf = lane >> 4;
    unsigned short* out = wpack + ((size_t)(c * 32 + lane)) * 16;
#pragma unroll
    for (int i = 0; i < 16; ++i) {
        int K = c * 32 + khalf * 16 + i;
        float v = (n < NTAG) ? W[(size_t)K * NTAG + n] : 0.0f;
        out[i] = f2bf(v);
    }
}

// ---------------------------------------------------------------------------
// Kernel 2: logits = log_softmax(leaky_relu(x) @ W + b).
// One wave per 16-token tile; bf16 WMMA, f32 accumulate; K-loop of 32 chunks.
// A staging: async global->LDS double buffer (CDNA5) or direct reg loads.
// ---------------------------------------------------------------------------
__global__ __launch_bounds__(256) void gemm_logsoftmax_kernel(
        const float* __restrict__ feat, const unsigned short* __restrict__ wpack,
        const float* __restrict__ bias, float* __restrict__ logits) {
    int lane = threadIdx.x & 31;
    int wave = threadIdx.x >> 5;
    int tile = blockIdx.x * 8 + wave;                 // 0..2047
    size_t token_base = (size_t)tile * 16;
    int m = lane & 15, khalf = lane >> 4;
    v8f c = {};

#if USE_ASYNC
    __shared__ float atile[8][2][512];                // per wave: 2 x 2KB f32 bufs
    // issue 4 async b128 copies for one 16x32 f32 chunk (coalesced rows)
#define ISSUE_CHUNK(KC)                                                        \
    {                                                                          \
        int _buf = (KC) & 1;                                                   \
        _Pragma("unroll")                                                      \
        for (int it = 0; it < 4; ++it) {                                       \
            int f4 = it * 32 + lane;                                           \
            int row = f4 >> 3;                                                 \
            int col = (f4 & 7) * 4;                                            \
            const float* g = feat + (token_base + row) * HID + (KC) * 32 + col;\
            __builtin_amdgcn_global_load_async_to_lds_b128(                    \
                (async_gptr)g,                                                 \
                (async_lptr)&atile[wave][_buf][f4 * 4],                        \
                0, 0);                                                         \
        }                                                                      \
    }
    ISSUE_CHUNK(0)
    for (int kc = 0; kc < 32; ++kc) {
        if (kc + 1 < 32) {
            ISSUE_CHUNK(kc + 1)
            WAIT_ASYNC(4);                            // chunk kc landed
        } else {
            WAIT_ASYNC(0);
        }
        const float* buf = &atile[wave][kc & 1][0];
        const float4 q0 = *(const float4*)(buf + m * 32 + khalf * 16);
        const float4 q1 = *(const float4*)(buf + m * 32 + khalf * 16 + 4);
        const float4 q2 = *(const float4*)(buf + m * 32 + khalf * 16 + 8);
        const float4 q3 = *(const float4*)(buf + m * 32 + khalf * 16 + 12);
#else
    __shared__ float ctile_s[8][256];
    const float* rowp = feat + (token_base + m) * HID;
    for (int kc = 0; kc < 32; ++kc) {
        // A fragment direct: K = kb+16*kh+0..15 (one 64B span)
        int k0 = kc * 32 + khalf * 16;
        const float4 q0 = *(const float4*)(rowp + k0);
        const float4 q1 = *(const float4*)(rowp + k0 + 4);
        const float4 q2 = *(const float4*)(rowp + k0 + 8);
        const float4 q3 = *(const float4*)(rowp + k0 + 12);
#endif
        BFrag a;
        a.v[0]  = (__bf16)lrelu(q0.x); a.v[1]  = (__bf16)lrelu(q0.y);
        a.v[2]  = (__bf16)lrelu(q0.z); a.v[3]  = (__bf16)lrelu(q0.w);
        a.v[4]  = (__bf16)lrelu(q1.x); a.v[5]  = (__bf16)lrelu(q1.y);
        a.v[6]  = (__bf16)lrelu(q1.z); a.v[7]  = (__bf16)lrelu(q1.w);
        a.v[8]  = (__bf16)lrelu(q2.x); a.v[9]  = (__bf16)lrelu(q2.y);
        a.v[10] = (__bf16)lrelu(q2.z); a.v[11] = (__bf16)lrelu(q2.w);
        a.v[12] = (__bf16)lrelu(q3.x); a.v[13] = (__bf16)lrelu(q3.y);
        a.v[14] = (__bf16)lrelu(q3.z); a.v[15] = (__bf16)lrelu(q3.w);
        BFrag b;
        const uint4* bp4 = (const uint4*)(wpack + ((size_t)(kc * 32 + lane)) * 16);
        b.q[0] = bp4[0];
        b.q[1] = bp4[1];
        c = __builtin_amdgcn_wmma_f32_16x16x32_bf16(false, a.v, false, b.v,
                                                    (short)0, c, false, false);
    }

    // C layout: VGPR r -> row M=r+8*khalf, lane&15 -> col n. Bias + LDS bounce.
#if USE_ASYNC
    float* cReg = &atile[wave][0][0];
#else
    float* cReg = &ctile_s[wave][0];
#endif
    int n = lane & 15;
    float bn = (n < NTAG) ? bias[n] : 0.0f;
#pragma unroll
    for (int r = 0; r < 8; ++r) {
        int M = r + 8 * khalf;
        cReg[M * 16 + n] = c[r] + bn;
    }
    // wave-local: per-wave LDS ops complete in order
    if (lane < 16) {
        float v[NTAG], mx = -3.0e30f;
#pragma unroll
        for (int j = 0; j < NTAG; ++j) { v[j] = cReg[lane * 16 + j]; mx = fmaxf(mx, v[j]); }
        float s = 0.0f;
#pragma unroll
        for (int j = 0; j < NTAG; ++j) s += __expf(v[j] - mx);
        float lse = mx + __logf(s);
        float* out = logits + (token_base + lane) * NTAG;
#pragma unroll
        for (int j = 0; j < NTAG; ++j) out[j] = v[j] - lse;
    }
}

// ---------------------------------------------------------------------------
// Kernel 3: CRF. One block per sequence; wave0 = forward+numerator,
// wave1 = Viterbi scan + LDS backtrack. States live on lanes 0..8.
// ---------------------------------------------------------------------------
__global__ __launch_bounds__(64) void crf_kernel(
        const float* __restrict__ logits, const int* __restrict__ mask_all,
        const int* __restrict__ labels_all, const float* __restrict__ trans,
        const float* __restrict__ start_t, const float* __restrict__ end_t,
        float* __restrict__ ll_ws, float* __restrict__ out_tags,
        float* __restrict__ out_scores) {
    __shared__ unsigned long long bpS[SEQ];           // 16KB backpointers
    int lane = threadIdx.x & 31;
    int wave = threadIdx.x >> 5;
    int b = blockIdx.x;
    const float* lb = logits + (size_t)b * SEQ * NTAG;
    const int* mk = mask_all + (size_t)b * SEQ;
    const int* lab = labels_all + (size_t)b * SEQ;
    int j = (lane < NTAG) ? lane : NTAG - 1;          // clamp idle lanes
    float stj = start_t[j], etj = end_t[j];
    float tr[NTAG];
#pragma unroll
    for (int i = 0; i < NTAG; ++i) tr[i] = trans[i * NTAG + j];

    if (wave == 0) {
        // ---- numerator: parallel over t across 32 lanes ----
        float se = 0.0f, st2 = 0.0f; int mc = 0;
        for (int t = lane; t < SEQ; t += 32) mc += (mk[t] != 0);
        for (int t = lane; t < SEQ - 1; t += 32) {
            if (mk[t]) se += lb[t * NTAG + lab[t]];
            if (mk[t + 1]) st2 += trans[lab[t] * NTAG + lab[t + 1]];
        }
        for (int off = 16; off; off >>= 1) {
            se  += __shfl_xor(se, off, 32);
            st2 += __shfl_xor(st2, off, 32);
            mc  += __shfl_xor(mc, off, 32);
        }
        // ---- forward (logsumexp) scan ----
        float alpha = stj + lb[j];
        for (int t = 1; t < SEQ; ++t) {
            float li = lb[t * NTAG + j];
            int mt = mk[t];
            float s[NTAG], mx = -3.0e30f;
#pragma unroll
            for (int i = 0; i < NTAG; ++i) {
                s[i] = __shfl(alpha, i, 32) + tr[i];
                mx = fmaxf(mx, s[i]);
            }
            float sum = 0.0f;
#pragma unroll
            for (int i = 0; i < NTAG; ++i) sum += __expf(s[i] - mx);
            float nv = mx + __logf(sum) + li;
            alpha = mt ? nv : alpha;
        }
        float fv = (lane < NTAG) ? alpha + etj : -3.0e30f;
        float mx2 = fv;
        for (int off = 16; off; off >>= 1) mx2 = fmaxf(mx2, __shfl_xor(mx2, off, 32));
        float sum2 = (lane < NTAG) ? __expf(fv - mx2) : 0.0f;
        for (int off = 16; off; off >>= 1) sum2 += __shfl_xor(sum2, off, 32);
        if (lane == 0) {
            float denom = mx2 + __logf(sum2);
            int last_idx = mc - 1;
            int last_tag = lab[last_idx];
            float num = start_t[lab[0]] + st2 + se + end_t[last_tag];
            if (mk[SEQ - 1]) num += lb[(SEQ - 1) * NTAG + last_tag];
            ll_ws[b] = num - denom;
        }
    } else {
        // ---- Viterbi with BIO constraints ----
        // allowed(i->j): j=='O'(0) or B-* (odd) => any; j = I-X (even>0)
        //   => only from B-X (j-1) or I-X (j).
        float tcv[NTAG];
#pragma unroll
        for (int i = 0; i < NTAG; ++i) {
            bool ok = (j == 0) || (j & 1) || (i == j - 1) || (i == j);
            tcv[i] = ok ? tr[i] : NEGC;
        }
        bool sok = (j == 0) || (j & 1);
        float av = (sok ? stj : NEGC) + lb[j];
        for (int t = 1; t < SEQ; ++t) {
            float li = lb[t * NTAG + j];
            int mt = mk[t];
            float best = -3.0e30f; int bpj = 0;
#pragma unroll
            for (int i = 0; i < NTAG; ++i) {
                float s = __shfl(av, i, 32) + tcv[i];
                if (s > best) { best = s; bpj = i; }   // first-max tie-break
            }
            av = mt ? best + li : av;
            int bpe = mt ? bpj : j;                    // identity row when masked
            unsigned long long u = 0ull;
#pragma unroll
            for (int i = 0; i < NTAG; ++i)
                u |= ((unsigned long long)(__shfl(bpe, i, 32) & 0xF)) << (4 * i);
            if (lane == 0) bpS[t] = u;
        }
        // final argmax (first occurrence on ties)
        float fv = (lane < NTAG) ? av + etj : -3.0e30f;
        int idx = lane;
        for (int off = 16; off; off >>= 1) {
            float ov = __shfl_xor(fv, off, 32);
            int oi = __shfl_xor(idx, off, 32);
            if (ov > fv || (ov == fv && oi < idx)) { fv = ov; idx = oi; }
        }
        if (lane == 0) {
            out_scores[b] = fv;
            int tag = idx;
            float* outT = out_tags + (size_t)b * SEQ;
            outT[SEQ - 1] = mk[SEQ - 1] ? (float)tag : 0.0f;
            for (int t = SEQ - 1; t >= 1; --t) {
                unsigned long long u = bpS[t];
                tag = (int)((u >> (4 * tag)) & 0xF);
                outT[t - 1] = mk[t - 1] ? (float)tag : 0.0f;
            }
        }
    }
}

// ---------------------------------------------------------------------------
// Kernel 4: loss = -sum(ll)/B  (single wave, deterministic)
// ---------------------------------------------------------------------------
__global__ void loss_kernel(const float* __restrict__ ll_ws, float* __restrict__ out) {
    int lane = threadIdx.x;
    float v = (lane < BATCH) ? ll_ws[lane] : 0.0f;
    for (int off = 16; off; off >>= 1) v += __shfl_xor(v, off, 32);
    if (lane == 0) out[0] = -v / (float)BATCH;
}

extern "C" void kernel_launch(void* const* d_in, const int* in_sizes, int n_in,
                              void* d_out, int out_size, void* d_ws, size_t ws_size,
                              hipStream_t stream) {
    const float* feat    = (const float*)d_in[0];
    const int*   mask    = (const int*)d_in[1];
    const int*   labels  = (const int*)d_in[2];
    const float* W       = (const float*)d_in[3];
    const float* bias    = (const float*)d_in[4];
    const float* trans   = (const float*)d_in[5];
    const float* start_t = (const float*)d_in[6];
    const float* end_t   = (const float*)d_in[7];
    float* out = (float*)d_out;

    char* ws = (char*)d_ws;
    unsigned short* wpack = (unsigned short*)ws;                 // 32 KB
    float* logits_ws = (float*)(ws + 64 * 1024);                 // 32768*9*4 = 1.125 MB
    float* ll_ws     = (float*)(ws + 64 * 1024 + 1179648);       // 16 floats

    hipLaunchKernelGGL(pack_w_kernel, dim3(32), dim3(32), 0, stream, W, wpack);
    hipLaunchKernelGGL(gemm_logsoftmax_kernel, dim3(256), dim3(256), 0, stream,
                       feat, wpack, bias, logits_ws);
    hipLaunchKernelGGL(crf_kernel, dim3(BATCH), dim3(64), 0, stream,
                       logits_ws, mask, labels, trans, start_t, end_t,
                       ll_ws, out + 1, out + 1 + BATCH * SEQ);
    hipLaunchKernelGGL(loss_kernel, dim3(1), dim3(32), 0, stream, ll_ws, out);
}
